// QVTree_1022202217004
// MI455X (gfx1250) — matrix operand order repaired and checked by reading, not compiled, not run on hardware
//
#include <hip/hip_runtime.h>
#include <hip/hip_bf16.h>
#include <math.h>

// ---------------------------------------------------------------------------
// QVTree scoring pipeline for MI455X (gfx1250, wave32, WMMA + TDM)
//   B=8, Lt=64, Lv=16384, D=256
//
// Roofline: v = 134MB read ONCE (row L2-norm folded into the WMMA K-loop and
// applied to the f32 accumulators post-GEMM), exp(S) staged as f16
// (16MB write + 16MB read). Total ~167MB @ 23.3TB/s ~= 7.2us; the 4.3 GFLOP
// f16 GEMM (v_wmma_f32_16x16x32_f16, K=256 -> 8 steps x 4 tiles per wave)
// is far below the WMMA ceiling -> memory bound, traffic-minimal.
//
// The per-WG t-tile (64x256 f16) is fetched by the Tensor Data Mover
// (tensor_load_to_lds) with hardware padding (pad every 512B row by 32B)
// producing the bank-conflict-free 272-half LDS stride directly.
//
// All softmax stability shifts are dropped (S in [-0.5,0.5], A in [0,1] ->
// shift-invariant & overflow-safe), so every cross-Lv reduction collapses
// into one fused atomic pass (LDS ds_add_f32 -> global_atomic_add_f32).
// ---------------------------------------------------------------------------

typedef _Float16 half16 __attribute__((ext_vector_type(16)));
typedef float    float8 __attribute__((ext_vector_type(8)));
typedef uint32_t u32x4  __attribute__((ext_vector_type(4)));
typedef uint32_t u32x8  __attribute__((ext_vector_type(8)));

#define B_    8
#define LT    64
#define LV    16384
#define D_    256
#define EPSF  1e-6f

// workspace layout (bytes)
#define TF16_OFF   ((size_t)0)
#define TF16_SZ    ((size_t)B_*LT*D_*2)                 // 256 KB  (f16 t, *0.5/||t||)
#define EXPS_OFF   (TF16_OFF + TF16_SZ)
#define EXPS_SZ    ((size_t)B_*LV*LT*2)                 // 16 MB   exp(S) as f16
#define STATS_OFF  (EXPS_OFF + EXPS_SZ)                 // E[512],Z[512],W[512],C[512]
#define STATS_SZ   ((size_t)4*512*4)
#define SCORES_OFF (STATS_OFF + STATS_SZ)
#define SCORES_SZ  ((size_t)B_*LV*4)                    // 512 KB
#define MM_OFF     (SCORES_OFF + SCORES_SZ)             // int min[8], max[8]

// f16-safe clamp; min/max "num" semantics also squash NaN to a finite value
// (the per-row 1/||v|| is applied to the f32 accumulators after the GEMM, so
// self-normalization keeps the math identical for all finite inputs).
__device__ __forceinline__ float san16(float x) {
  return fminf(fmaxf(x, -65504.f), 65504.f);
}
__device__ __forceinline__ float san(float x) {
  if (x != x) return 0.f;
  return fminf(fmaxf(x, -3.402823466e38f), 3.402823466e38f);
}

// --------------------------- init accumulators ------------------------------
__global__ void k_init(float* __restrict__ stats, int* __restrict__ mm) {
  int i = blockIdx.x * blockDim.x + threadIdx.x;
  if (i < 2048) stats[i] = 0.f;            // E, Z, W, C
  if (i < 8)             mm[i]     = 0x7F7FFFFF;  // min init = FLT_MAX bits
  else if (i < 16)       mm[i]     = 0;           // max init = 0 (scores >= 0)
}

// ------------------- normalize t, fold 1/TEMP, convert f16 ------------------
__global__ void k_tnorm(const float* __restrict__ t, _Float16* __restrict__ tf) {
  int lane = threadIdx.x & 31;
  int row  = blockIdx.x * 4 + (threadIdx.x >> 5);   // 512 rows total
  const float* rp = t + (size_t)row * D_ + lane * 8;
  float4 x0 = *reinterpret_cast<const float4*>(rp);
  float4 x1 = *reinterpret_cast<const float4*>(rp + 4);
  float e[8];
  e[0]=san(x0.x); e[1]=san(x0.y); e[2]=san(x0.z); e[3]=san(x0.w);
  e[4]=san(x1.x); e[5]=san(x1.y); e[6]=san(x1.z); e[7]=san(x1.w);
  float ss = 0.f;
  #pragma unroll
  for (int i = 0; i < 8; i++) ss += e[i] * e[i];
  ss += __shfl_xor(ss, 1, 32);  ss += __shfl_xor(ss, 2, 32);
  ss += __shfl_xor(ss, 4, 32);  ss += __shfl_xor(ss, 8, 32);
  ss += __shfl_xor(ss, 16, 32);
  float scale = 0.5f / fmaxf(sqrtf(ss), EPSF);      // 0.5 = 1/TEMP folded in
  __align__(16) _Float16 h[8];
  #pragma unroll
  for (int i = 0; i < 8; i++) h[i] = (_Float16)(e[i] * scale);
  *reinterpret_cast<uint4*>(tf + (size_t)row * D_ + lane * 8) =
      *reinterpret_cast<uint4*>(h);
}

// ---- main: single-pass WMMA on raw v, post-GEMM row scaling, exp(S) --------
// grid: 1024 blocks (8 batches x 128 chunks), 256 threads = 8 waves.
// Each wave computes a 16(v) x 64(t) strip of S with 4 accumulators.
__launch_bounds__(256, 2)
__global__ void k_main(const float* __restrict__ v, const _Float16* __restrict__ tf,
                       _Float16* __restrict__ expH, float* __restrict__ gE,
                       float* __restrict__ gZ,     float* __restrict__ gW) {
  constexpr int TSTR = 272;                      // padded row stride (halves)
  __shared__ __align__(32) _Float16 tlds[64 * TSTR];
  __shared__ float ldsE[64], ldsZ[64], ldsW[64];

  int tid   = threadIdx.x;
  int b     = blockIdx.x >> 7;
  int chunk = blockIdx.x & 127;
  int lane  = tid & 31;
  int wave  = tid >> 5;

  // --- TDM: DMA this batch's t (64 rows x 512B) into LDS with HW padding ---
  // D# per ISA 8.3/8.4: group0 = {count=1, lds_addr, global_addr(57b), type=2}
  // group1: data_size=2B; pad_enable, pad_interval=6 (every 128 DW = 512B row),
  // pad_amount=7 (8 DW = 32B) -> LDS row stride 544B = 272 halves;
  // tensor_dim0=256, tensor_dim1=64, tile 256x64, dim0_stride=256.
  if (wave == 0) {
    uint64_t ga = (uint64_t)(uintptr_t)(tf + (size_t)b * 64 * D_);
    uint32_t la = (uint32_t)(uintptr_t)(&tlds[0]);   // generic low 32b = LDS addr
    u32x4 g0;
    g0[0] = 1u;                                      // count=1 (valid D#)
    g0[1] = la;                                      // lds_addr
    g0[2] = (uint32_t)ga;                            // global_addr[31:0]
    g0[3] = (uint32_t)((ga >> 32) & 0x01FFFFFFu) | (2u << 30);  // addr[56:32]|type=2
    u32x8 g1;
    g1[0] = (1u << 16) | (1u << 20) | (6u << 22) | (7u << 25);  // ds=2B,pad
    g1[1] = (256u & 0xFFFFu) << 16;                  // tensor_dim0[15:0]
    g1[2] = (64u  & 0xFFFFu) << 16;                  // tensor_dim1[15:0]
    g1[3] = (256u & 0xFFFFu) << 16;                  // tile_dim0
    g1[4] = 64u;                                     // tile_dim1 (tile_dim2=0)
    g1[5] = 256u;                                    // tensor_dim0_stride[31:0]
    g1[6] = 0u;
    g1[7] = 0u;
    u32x4 gz = {0u, 0u, 0u, 0u};                     // groups 2/3 unused (2D)
    asm volatile("tensor_load_to_lds %0, %1, %2, %3"
                 :: "s"(g0), "s"(g1), "s"(gz), "s"(gz) : "memory");
    __builtin_amdgcn_s_wait_tensorcnt(0);
  }
  if (tid < 64) { ldsE[tid] = 0.f; ldsZ[tid] = 0.f; ldsW[tid] = 0.f; }
  __syncthreads();

  int rowbase = chunk * 128 + wave * 16;
  int m16 = lane & 15;                           // A-row / B-col / C-col index
  int hi  = lane >> 4;                           // upper-half-wave flag
  const float* arow = v + ((size_t)b * LV + rowbase + m16) * D_;

  float8 acc[4];
  #pragma unroll
  for (int tt = 0; tt < 4; tt++)
    #pragma unroll
    for (int i = 0; i < 8; i++) acc[tt][i] = 0.f;

  // K loop: 8 steps of 32. A-frag lane layout (contiguous k, same order as B-frag):
  //   lane L: M = L&15; halves 0..15 -> K = kb+16*hi+{0..15}
  // Sum-of-squares of the loaded elements is accumulated in the same pass;
  // lanes L and L+16 together cover row (L&15) exactly once over the K loop.
  float ssp = 0.f;
  #pragma unroll
  for (int kk = 0; kk < 8; kk++) {
    int kb = kk * 32 + hi * 16;
    const float4* p0 = reinterpret_cast<const float4*>(arow + kb);
    const float4* p1 = reinterpret_cast<const float4*>(arow + kb + 8);
    float4 c0 = p0[0], c1 = p0[1], c2 = p1[0], c3 = p1[1];
    if (kk < 7) __builtin_prefetch(arow + kb + 32, 0, 1);  // global_prefetch_b8
    float s0 = san16(c0.x), s1 = san16(c0.y), s2 = san16(c0.z), s3 = san16(c0.w);
    float s4 = san16(c1.x), s5 = san16(c1.y), s6 = san16(c1.z), s7 = san16(c1.w);
    float s8 = san16(c2.x), s9 = san16(c2.y), sa = san16(c2.z), sb = san16(c2.w);
    float sc = san16(c3.x), sd = san16(c3.y), se = san16(c3.z), sf = san16(c3.w);
    ssp += s0*s0 + s1*s1 + s2*s2 + s3*s3;
    ssp += s4*s4 + s5*s5 + s6*s6 + s7*s7;
    ssp += s8*s8 + s9*s9 + sa*sa + sb*sb;
    ssp += sc*sc + sd*sd + se*se + sf*sf;
    half16 a;
    a[0]=(_Float16)s0;  a[1]=(_Float16)s1;  a[2]=(_Float16)s2;  a[3]=(_Float16)s3;
    a[4]=(_Float16)s4;  a[5]=(_Float16)s5;  a[6]=(_Float16)s6;  a[7]=(_Float16)s7;
    a[8]=(_Float16)s8;  a[9]=(_Float16)s9;  a[10]=(_Float16)sa; a[11]=(_Float16)sb;
    a[12]=(_Float16)sc; a[13]=(_Float16)sd; a[14]=(_Float16)se; a[15]=(_Float16)sf;

    // B-frag (32x16): lane L -> N = L&15, halves j -> K = koff + j
    int koff = kk * 32 + hi * 16;
    #pragma unroll
    for (int tt = 0; tt < 4; tt++) {
      half16 bf = *reinterpret_cast<const half16*>(
          &tlds[(size_t)(tt * 16 + m16) * TSTR + koff]);
      acc[tt] = __builtin_amdgcn_wmma_f32_16x16x32_f16(
          false, a, false, bf, (short)0, acc[tt], false, false);
    }
  }

  // row scales: lane L holds ss of row (L&15) after xor16; each lane needs
  // scales of rows r+8*hi -> one variable-source shuffle per accumulator reg.
  ssp += __shfl_xor(ssp, 16, 32);
  float scale = 1.f / fmaxf(sqrtf(ssp), EPSF);
  float scl[8];
  #pragma unroll
  for (int r = 0; r < 8; r++) scl[r] = __shfl(scale, r + (hi << 3), 32);

  // acc*scl = S (already /TEMP). C/D layout: VGPR r, lane L -> (M=r+8*hi, N=L&15)
  // Row softmax over the 64 t values: e = exp(S); rowsum across 16 lanes+4 tiles.
  float rs[8];
  #pragma unroll
  for (int r = 0; r < 8; r++) rs[r] = 0.f;
  #pragma unroll
  for (int tt = 0; tt < 4; tt++)
    #pragma unroll
    for (int r = 0; r < 8; r++) {
      float e = __expf(acc[tt][r] * scl[r]);
      acc[tt][r] = e;                     // acc now holds exp(S)
      rs[r] += e;
    }
  #pragma unroll
  for (int r = 0; r < 8; r++) {
    rs[r] += __shfl_xor(rs[r], 1, 32);
    rs[r] += __shfl_xor(rs[r], 2, 32);
    rs[r] += __shfl_xor(rs[r], 4, 32);
    rs[r] += __shfl_xor(rs[r], 8, 32);
    rs[r] = 1.f / rs[r];
  }

  // store exp(S) as f16 [B][LV][LT]  (range (0.61, 1.65) -> f16-exact to ~1e-4)
  #pragma unroll
  for (int tt = 0; tt < 4; tt++) {
    int tcol = tt * 16 + m16;
    #pragma unroll
    for (int r = 0; r < 8; r++) {
      int vr = rowbase + r + hi * 8;
      expH[((size_t)b * LV + vr) * LT + tcol] = (_Float16)acc[tt][r];
    }
  }

  // fused per-t (column) stats over this strip's 16 rows:
  //   E += exp(S);  Z += exp(A/2);  W += exp(A/2)*A   with A = exp(S)/rowsum
  #pragma unroll
  for (int tt = 0; tt < 4; tt++) {
    float esum = 0.f, za = 0.f, wa = 0.f;
    #pragma unroll
    for (int r = 0; r < 8; r++) {
      float e = acc[tt][r];
      float A = e * rs[r];
      float q = __expf(0.5f * A);
      esum += e; za += q; wa += q * A;
    }
    esum += __shfl_xor(esum, 16, 32);
    za   += __shfl_xor(za,   16, 32);
    wa   += __shfl_xor(wa,   16, 32);
    if (hi == 0) {
      int t = tt * 16 + m16;
      atomicAdd(&ldsE[t], esum);          // ds_add_f32
      atomicAdd(&ldsZ[t], za);
      atomicAdd(&ldsW[t], wa);
    }
  }
  __syncthreads();
  if (tid < 64) {                          // one global atomic per t per WG
    atomicAdd(&gE[b * 64 + tid], ldsE[tid]);
    atomicAdd(&gZ[b * 64 + tid], ldsZ[tid]);
    atomicAdd(&gW[b * 64 + tid], ldsW[tid]);
  }
}

// ------------------ text_score + per-t scale c = ts / E ---------------------
__global__ void k_textscore(const float* __restrict__ gE, const float* __restrict__ gZ,
                            const float* __restrict__ gW, float* __restrict__ gC) {
  __shared__ float red[64];
  int b = blockIdx.x, t = threadIdx.x;
  float ts = gW[b * 64 + t] / gZ[b * 64 + t];   // = sum(weights*A) over v
  red[t] = ts; __syncthreads();
  for (int s = 32; s > 0; s >>= 1) { if (t < s) red[t] += red[t + s]; __syncthreads(); }
  float tsn = ts / (red[0] + EPSF);
  gC[b * 64 + t] = tsn / gE[b * 64 + t];
}

// --------------- scores[b,v] = exp(S) row . c[b]  + block min/max -----------
__global__ void k_scores(const _Float16* __restrict__ expH, const float* __restrict__ gC,
                         float* __restrict__ scores, int* __restrict__ mm) {
  __shared__ __align__(16) float cl[64];
  __shared__ float rmin[256], rmax[256];
  int b = blockIdx.x >> 6, blk = blockIdx.x & 63;
  int tid = threadIdx.x;
  if (tid < 64) cl[tid] = gC[b * 64 + tid];
  __syncthreads();
  int vr = blk * 256 + tid;
  const half16* row = reinterpret_cast<const half16*>(expH + ((size_t)b * LV + vr) * LT);
  float acc = 0.f;
  #pragma unroll
  for (int i = 0; i < 4; i++) {
    half16 h = row[i];
    #pragma unroll
    for (int j = 0; j < 16; j++) acc = fmaf((float)h[j], cl[i * 16 + j], acc);
  }
  scores[(size_t)b * LV + vr] = acc;
  rmin[tid] = acc; rmax[tid] = acc; __syncthreads();
  for (int s = 128; s > 0; s >>= 1) {
    if (tid < s) {
      rmin[tid] = fminf(rmin[tid], rmin[tid + s]);
      rmax[tid] = fmaxf(rmax[tid], rmax[tid + s]);
    }
    __syncthreads();
  }
  if (tid == 0) {   // scores >= 0 -> int-bit compare order is valid
    atomicMin(&mm[b],     __float_as_int(rmin[0]));
    atomicMax(&mm[8 + b], __float_as_int(rmax[0]));
  }
}

// ------------------------- min-max normalize to out -------------------------
__global__ void k_final(const float* __restrict__ scores, const int* __restrict__ mm,
                        float* __restrict__ out) {
  int i = blockIdx.x * 256 + threadIdx.x;
  int b = i >> 14;
  float mn = __int_as_float(mm[b]);
  float mx = __int_as_float(mm[8 + b]);
  out[i] = (scores[i] - mn) / (mx - mn + EPSF);
}

// ---------------------------------------------------------------------------
extern "C" void kernel_launch(void* const* d_in, const int* in_sizes, int n_in,
                              void* d_out, int out_size, void* d_ws, size_t ws_size,
                              hipStream_t stream) {
  const float* t = (const float*)d_in[0];
  const float* v = (const float*)d_in[1];
  char* ws = (char*)d_ws;
  _Float16* tf   = (_Float16*)(ws + TF16_OFF);
  _Float16* expH = (_Float16*)(ws + EXPS_OFF);
  float* gE      = (float*)(ws + STATS_OFF);
  float* gZ      = gE + 512;
  float* gW      = gZ + 512;
  float* gC      = gW + 512;
  float* scores  = (float*)(ws + SCORES_OFF);
  int*   mm      = (int*)(ws + MM_OFF);

  k_init     <<<8,    256, 0, stream>>>(gE, mm);
  k_tnorm    <<<128,  128, 0, stream>>>(t, tf);
  k_main     <<<1024, 256, 0, stream>>>(v, tf, expH, gE, gZ, gW);
  k_textscore<<<8,    64,  0, stream>>>(gE, gZ, gW, gC);
  k_scores   <<<512,  256, 0, stream>>>(expH, gC, scores, mm);
  k_final    <<<512,  256, 0, stream>>>(scores, mm, (float*)d_out);
}
